// WindowAttention_12240656794299
// MI455X (gfx1250) — matrix operand order repaired and checked by reading, not compiled, not run on hardware
//
#include <hip/hip_runtime.h>
#include <hip/hip_bf16.h>

// ---------------- types for CDNA5 WMMA ----------------
typedef __attribute__((ext_vector_type(16))) __bf16 v16bf;
typedef __attribute__((ext_vector_type(8)))  float  v8f;

union ABFrag { v16bf v; uint4 u[2]; };

__device__ __forceinline__ unsigned short f32_to_bf16(float f) {
    unsigned int u = __float_as_uint(f);
    u += 0x7FFFu + ((u >> 16) & 1u);   // round-to-nearest-even
    return (unsigned short)(u >> 16);
}

__device__ __forceinline__ unsigned int pack_bf16x2(float lo, float hi) {
    return (unsigned int)f32_to_bf16(lo) | ((unsigned int)f32_to_bf16(hi) << 16);
}

// A-matrix 16x32 bf16 fragment from row-major storage (LDS or global).
__device__ __forceinline__ v16bf load_a_frag(const unsigned short* base, int row_stride,
                                             int m_base, int k_base, int lane) {
    const unsigned short* p = base + (m_base + (lane & 15)) * row_stride
                                   + k_base + ((lane >> 4) << 3);
    ABFrag f;
    f.u[0] = *(const uint4*)(p);
    f.u[1] = *(const uint4*)(p + 16);
    return f.v;
}

// B-matrix 32x16 bf16 fragment; source column-major (column n contiguous in K).
__device__ __forceinline__ v16bf load_b_frag(const unsigned short* colbase, int col_stride,
                                             int n_base, int k_base, int lane) {
    const unsigned short* p = colbase + (n_base + (lane & 15)) * col_stride
                                      + k_base + ((lane >> 4) << 3);
    ABFrag f;
    f.u[0] = *(const uint4*)(p);
    f.u[1] = *(const uint4*)(p + 16);
    return f.v;
}

__device__ __forceinline__ v8f wmma_bf16(v16bf a, v16bf b, v8f c) {
    return __builtin_amdgcn_wmma_f32_16x16x32_bf16(
        false, a, false, b, (short)0, c, false, false);
}

// ---------------- problem constants ----------------
#define NWINDOWS 4096
#define NTOK     64
#define CDIM     192
#define HEADS    6
#define HD       32
#define ATT_SCALE 0.17677669529663687f   // 32^-0.5
#define LOG2E     1.4426950408889634f

// LDS layout (ushort units). stride_bytes % 32 == 16 -> conflict-free b128.
#define XS_STRIDE 200
#define QK_STRIDE 40
#define VT_STRIDE 72
#define AT_STRIDE 72

#define XS_OFF  0
#define QS_OFF  (XS_OFF + NTOK * XS_STRIDE)
#define KS_OFF  (QS_OFF + HEADS * NTOK * QK_STRIDE)
#define VT_OFF  (KS_OFF + HEADS * NTOK * QK_STRIDE)
#define AT_OFF  (VT_OFF + HEADS * HD * VT_STRIDE)
#define LDS_USHORTS (AT_OFF + HEADS * NTOK * AT_STRIDE)

// ---------------- prep 1: bf16 weights + gathered rel-pos bias ----------------
__global__ void winattn_prep_kernel(const float* __restrict__ qkv_w,
                                    const float* __restrict__ proj_w,
                                    const float* __restrict__ table,
                                    unsigned short* __restrict__ qkvw_bf,
                                    unsigned short* __restrict__ projw_bf,
                                    float* __restrict__ bias) {
    int i = blockIdx.x * blockDim.x + threadIdx.x;
    if (i < 3 * CDIM * CDIM) qkvw_bf[i] = f32_to_bf16(qkv_w[i]);
    if (i < CDIM * CDIM)     projw_bf[i] = f32_to_bf16(proj_w[i]);
    if (i < HEADS * NTOK * NTOK) {
        int h = i / (NTOK * NTOK);
        int rem = i - h * NTOK * NTOK;
        int r = rem >> 6, c = rem & 63;
        int idx = ((r >> 3) - (c >> 3) + 7) * 15 + ((r & 7) - (c & 7) + 7);
        bias[i] = table[idx * HEADS + h];
    }
}

// ---------------- prep 2: combined bias+mask table bm[64][6][64][64] --------
__global__ void winattn_bm_kernel(const float* __restrict__ bias,
                                  const float* __restrict__ mask,
                                  float* __restrict__ bm) {
    int i = blockIdx.x * blockDim.x + threadIdx.x;   // < 64*6*4096
    int ij  = i & 4095;
    int h   = (i >> 12) % HEADS;
    int wid = i / (HEADS * 4096);
    bm[i] = bias[h * 4096 + ij] + mask[wid * 4096 + ij];
}

// ---------------- main: one workgroup (8 wave32) per window ----------------
__global__ __launch_bounds__(256)
void winattn_main_kernel(const float* __restrict__ x,
                         const float* __restrict__ mask,
                         const float* __restrict__ qkv_b,
                         const float* __restrict__ proj_b,
                         const unsigned short* __restrict__ qkvw_bf,
                         const unsigned short* __restrict__ projw_bf,
                         const float* __restrict__ bias,
                         const float* __restrict__ bm,
                         int use_bm,
                         float* __restrict__ out,
                         float* __restrict__ attn_out) {
    extern __shared__ unsigned short smem[];
    unsigned short* xs  = smem + XS_OFF;
    unsigned short* qs  = smem + QS_OFF;
    unsigned short* ksm = smem + KS_OFF;
    unsigned short* vt  = smem + VT_OFF;
    unsigned short* at  = smem + AT_OFF;

    const int b    = blockIdx.x;
    const int tid  = threadIdx.x;
    // force wave-uniformity so per-tile control flow becomes scalar branches
    const int wave = __builtin_amdgcn_readfirstlane(tid >> 5);
    const int lane = tid & 31;
    const int half = lane >> 4;
    const int lc   = lane & 15;

    // ---- load x tile, convert to bf16 ----
    const float* xb = x + (size_t)b * NTOK * CDIM;
    #pragma unroll 4
    for (int i = tid; i < (NTOK * CDIM) / 4; i += 256) {
        float4 f = ((const float4*)xb)[i];
        int e = i << 2;
        unsigned int* d = (unsigned int*)(xs + (e / CDIM) * XS_STRIDE + (e % CDIM));
        d[0] = pack_bf16x2(f.x, f.y);
        d[1] = pack_bf16x2(f.z, f.w);
    }
    __syncthreads();

    // ---- stage A: qkv = x @ qkv_w^T + qkv_b  (wave owns M-strip mt) ----
    {
        const int mt  = wave & 3;
        const int ntb = wave >> 2;          // 0 or 1
        v16bf afr[6];
        #pragma unroll
        for (int ks = 0; ks < 6; ++ks)
            afr[ks] = load_a_frag(xs, XS_STRIDE, mt * 16, ks * 32, lane);

        for (int i = 0; i < 18; ++i) {
            const int nt = ntb + 2 * i;     // uniform, 0..35
            v8f c = {};
            #pragma unroll
            for (int ks = 0; ks < 6; ++ks)
                c = wmma_bf16(afr[ks],
                              load_b_frag(qkvw_bf, CDIM, nt * 16, ks * 32, lane), c);
            const float bv = qkv_b[nt * 16 + lc];
            if (nt < 12) {                  // ---- q (scaled) ----
                const int h = nt >> 1, d = ((nt & 1) << 4) + lc;
                #pragma unroll
                for (int r = 0; r < 8; ++r) {
                    int row = mt * 16 + (half << 3) + r;
                    qs[(h * NTOK + row) * QK_STRIDE + d] =
                        f32_to_bf16((c[r] + bv) * ATT_SCALE);
                }
            } else if (nt < 24) {           // ---- k ----
                const int p = nt - 12, h = p >> 1, d = ((p & 1) << 4) + lc;
                #pragma unroll
                for (int r = 0; r < 8; ++r) {
                    int row = mt * 16 + (half << 3) + r;
                    ksm[(h * NTOK + row) * QK_STRIDE + d] = f32_to_bf16(c[r] + bv);
                }
            } else {                        // ---- v (transposed, packed b128) ----
                const int p = nt - 24, h = p >> 1, d = ((p & 1) << 4) + lc;
                uint4 pk;
                pk.x = pack_bf16x2(c[0] + bv, c[1] + bv);
                pk.y = pack_bf16x2(c[2] + bv, c[3] + bv);
                pk.z = pack_bf16x2(c[4] + bv, c[5] + bv);
                pk.w = pack_bf16x2(c[6] + bv, c[7] + bv);
                *(uint4*)(vt + (h * HD + d) * VT_STRIDE + mt * 16 + (half << 3)) = pk;
            }
        }
    }
    __syncthreads();

    // ---- stage B: attn = softmax(q k^T + bias + mask); 24 row-strips ----
    const int wid = b & 63;
    for (int s = wave; s < 24; s += 8) {
        const int h = s >> 2, it = s & 3;
        v16bf a = load_a_frag(qs + h * NTOK * QK_STRIDE, QK_STRIDE, it * 16, 0, lane);
        v8f cc[4];
        #pragma unroll
        for (int jt = 0; jt < 4; ++jt) {
            v8f c = {};
            cc[jt] = wmma_bf16(a,
                load_b_frag(ksm + h * NTOK * QK_STRIDE, QK_STRIDE, jt * 16, 0, lane), c);
        }
        const float* bmp = bm + (((size_t)wid * HEADS + h) * NTOK) * NTOK;
        const float* bp  = bias + h * NTOK * NTOK;
        const float* mp  = mask + (size_t)wid * NTOK * NTOK;
        float* ao = attn_out + (((size_t)b * HEADS + h) * NTOK) * NTOK;
        unsigned short* as = at + h * NTOK * AT_STRIDE;
        #pragma unroll
        for (int r = 0; r < 8; ++r) {
            const int i = it * 16 + (half << 3) + r;
            float mx = -3.402823e38f;
            #pragma unroll
            for (int jt = 0; jt < 4; ++jt) {
                int j = jt * 16 + lc;
                float add = use_bm ? bmp[i * NTOK + j]
                                   : bp[i * NTOK + j] + mp[i * NTOK + j];
                cc[jt][r] += add;
                mx = fmaxf(mx, cc[jt][r]);
            }
            #pragma unroll
            for (int off = 1; off < 16; off <<= 1)
                mx = fmaxf(mx, __shfl_xor(mx, off, 32));
            float sum = 0.f;
            #pragma unroll
            for (int jt = 0; jt < 4; ++jt) {
                float e = exp2f((cc[jt][r] - mx) * LOG2E);
                cc[jt][r] = e;
                sum += e;
            }
            #pragma unroll
            for (int off = 1; off < 16; off <<= 1)
                sum += __shfl_xor(sum, off, 32);
            const float inv = 1.0f / sum;
            #pragma unroll
            for (int jt = 0; jt < 4; ++jt) {
                int j = jt * 16 + lc;
                float p = cc[jt][r] * inv;
                ao[i * NTOK + j] = p;                     // fp32 attn output
                as[i * AT_STRIDE + j] = f32_to_bf16(p);   // PV GEMM operand
            }
        }
    }
    __syncthreads();

    // ---- stage C: o = attn @ v ; A-frags shared across the two d-tiles ----
    for (int s = wave; s < 24; s += 8) {
        const int h = s >> 2, it = s & 3;
        v16bf a0 = load_a_frag(at + h * NTOK * AT_STRIDE, AT_STRIDE, it * 16,  0, lane);
        v16bf a1 = load_a_frag(at + h * NTOK * AT_STRIDE, AT_STRIDE, it * 16, 32, lane);
        #pragma unroll
        for (int dt = 0; dt < 2; ++dt) {
            v8f c = {};
            c = wmma_bf16(a0, load_b_frag(vt + h * HD * VT_STRIDE, VT_STRIDE,
                                          dt * 16,  0, lane), c);
            c = wmma_bf16(a1, load_b_frag(vt + h * HD * VT_STRIDE, VT_STRIDE,
                                          dt * 16, 32, lane), c);
            const int dcol = h * HD + dt * 16 + lc;
            #pragma unroll
            for (int r = 0; r < 8; ++r) {
                int i = it * 16 + (half << 3) + r;
                xs[i * XS_STRIDE + dcol] = f32_to_bf16(c[r]);
            }
        }
    }
    __syncthreads();

    // ---- stage D: out = o @ proj_w^T + proj_b (wave owns M-strip mt) ----
    {
        const int mt  = wave & 3;
        const int ntb = wave >> 2;
        v16bf afr[6];
        #pragma unroll
        for (int ks = 0; ks < 6; ++ks)
            afr[ks] = load_a_frag(xs, XS_STRIDE, mt * 16, ks * 32, lane);
        float* ob = out + (size_t)b * NTOK * CDIM;
        for (int i = 0; i < 6; ++i) {
            const int nt = ntb + 2 * i;     // 0..11
            v8f c = {};
            #pragma unroll
            for (int ks = 0; ks < 6; ++ks)
                c = wmma_bf16(afr[ks],
                              load_b_frag(projw_bf, CDIM, nt * 16, ks * 32, lane), c);
            const int o = nt * 16 + lc;
            const float bv = proj_b[o];
            #pragma unroll
            for (int r = 0; r < 8; ++r) {
                int row = mt * 16 + (half << 3) + r;
                ob[row * CDIM + o] = c[r] + bv;
            }
        }
    }
}

extern "C" void kernel_launch(void* const* d_in, const int* in_sizes, int n_in,
                              void* d_out, int out_size, void* d_ws, size_t ws_size,
                              hipStream_t stream) {
    const float* x      = (const float*)d_in[0];
    const float* mask   = (const float*)d_in[1];
    const float* qkv_w  = (const float*)d_in[2];
    const float* qkv_b  = (const float*)d_in[3];
    const float* proj_w = (const float*)d_in[4];
    const float* proj_b = (const float*)d_in[5];
    const float* table  = (const float*)d_in[6];

    float* out  = (float*)d_out;
    float* attn = out + (size_t)NWINDOWS * NTOK * CDIM;

    unsigned short* qkvw_bf  = (unsigned short*)d_ws;              // 576*192 bf16
    unsigned short* projw_bf = qkvw_bf + 3 * CDIM * CDIM;          // 192*192 bf16
    float*          bias     = (float*)(projw_bf + CDIM * CDIM);   // 6*64*64 f32
    float*          bm       = bias + HEADS * NTOK * NTOK;         // 64*6*64*64 f32

    const size_t base_bytes = (size_t)(3 * CDIM * CDIM + CDIM * CDIM) * 2
                            + (size_t)HEADS * NTOK * NTOK * 4;
    const size_t bm_bytes   = (size_t)64 * HEADS * NTOK * NTOK * 4;
    const int use_bm = (ws_size >= base_bytes + bm_bytes) ? 1 : 0;

    int prep_n = 3 * CDIM * CDIM;
    winattn_prep_kernel<<<(prep_n + 255) / 256, 256, 0, stream>>>(
        qkv_w, proj_w, table, qkvw_bf, projw_bf, bias);

    if (use_bm) {
        int bm_n = 64 * HEADS * NTOK * NTOK;
        winattn_bm_kernel<<<bm_n / 256, 256, 0, stream>>>(bias, mask, bm);
    }

    size_t lds_bytes = (size_t)LDS_USHORTS * sizeof(unsigned short);   // ~166 KB
    winattn_main_kernel<<<NWINDOWS, 256, lds_bytes, stream>>>(
        x, mask, qkv_b, proj_b, qkvw_bf, projw_bf, bias, bm, use_bm, out, attn);
}
